// Attention_13778255085887
// MI455X (gfx1250) — matrix operand order repaired and checked by reading, not compiled, not run on hardware
//
#include <hip/hip_runtime.h>
#include <hip/hip_bf16.h>

// ---------------------------------------------------------------------------
// Attention block on gfx1250 (MI455X).
//   Matmuls: v_wmma_f32_16x16x32_f16.  GEMM operand staging: TDM
//   (tensor_load_to_lds + s_wait_tensorcnt), double-buffered against WMMA.
//   B=8, dim=256, L=1024, NH=8, KEY_DIM=16, HEAD_DIM=32
// ---------------------------------------------------------------------------

typedef __attribute__((ext_vector_type(16))) _Float16 v16h;
typedef __attribute__((ext_vector_type(8)))  float    v8f;
typedef __attribute__((ext_vector_type(4)))  unsigned int v4u;
typedef __attribute__((ext_vector_type(8)))  int      v8i;
typedef __attribute__((ext_vector_type(4)))  int      v4i;

#define B_   8
#define DIM_ 256
#define L_   1024
#define NH_  8
#define KD_  16
#define HD_  32
#define H_   512          // qkv rows = NH*(2*KD+HD)
#define SCALE_ 0.25f      // KEY_DIM^-0.5

__device__ __forceinline__ v8f wmma16(v16h a, v16h b, v8f c) {
  return __builtin_amdgcn_wmma_f32_16x16x32_f16(false, a, false, b, (short)0, c,
                                                false, false);
}

// LDS aperture: generic pointers to __shared__ carry the LDS byte offset in
// their low 32 bits (ISA 10.2).
__device__ __forceinline__ unsigned lds_off_u32(const void* p) {
  return (unsigned)(unsigned long long)p;
}

// ---------------------------------------------------------------------------
// TDM: issue a 2D tile load (tile_h rows x tile_w elements, row stride in
// elements) from global into LDS, with hardware row padding.
// D# layout per CDNA5 ISA section 8 (groups 0/1; groups 2/3 zero for 2D).
// esize_code: 0=1B,1=2B,2=4B.  pad codes per ISA (interval: 2^(k+1) DWORDs,
// amount: k+1 DWORDs).  This toolchain's builtin takes 6 args
// (g0, g1, g2, g3, extra, cpol); higher groups are unused for 2D tiles.
// ---------------------------------------------------------------------------
__device__ __forceinline__ void tdm_load_2d(unsigned lds_byte_off,
                                            const void* gaddr, unsigned tile_w,
                                            unsigned tile_h,
                                            unsigned long long row_stride_elems,
                                            unsigned esize_code,
                                            unsigned pad_interval_code,
                                            unsigned pad_amount_code) {
  unsigned long long ga = (unsigned long long)gaddr;
  v4u g0;
  g0[0] = 1u;                                         // count=1, no gather
  g0[1] = lds_byte_off;                               // lds_addr [63:32]
  g0[2] = (unsigned)(ga & 0xffffffffu);               // global_addr lo
  g0[3] = (unsigned)((ga >> 32) & 0x01ffffffu)        // global_addr hi [120:96]
          | (2u << 30);                               // type=2 ("image")
  v8i g1;
  g1[0] = (int)((esize_code << 16) | (1u << 20)       // data_size, pad_enable
                | (pad_interval_code << 22) | (pad_amount_code << 25));
  g1[1] = (int)(tile_w << 16);                        // tensor_dim0[15:0]
  g1[2] = (int)((tile_w >> 16) | (tile_h << 16));     // dim0 hi | dim1 lo
  g1[3] = (int)((tile_h >> 16) | (tile_w << 16));     // dim1 hi | tile_dim0
  g1[4] = (int)(tile_h & 0xffffu);                    // tile_dim1 (tile_dim2=0)
  g1[5] = (int)(unsigned)(row_stride_elems & 0xffffffffu);   // dim0_stride lo
  g1[6] = (int)(unsigned)((row_stride_elems >> 32) & 0xffffu);
  g1[7] = 0;
  v4i z4 = {0, 0, 0, 0};
  v8i z8 = {0, 0, 0, 0, 0, 0, 0, 0};
  __builtin_amdgcn_tensor_load_to_lds(g0, g1, z4, z4, z8, 0);
}

// ---------------------------------------------------------------------------
// Generic bias-GEMM:  Y[b, o, l] = sum_d W[o,d] * X[b,d,l] + bias[o]
// 256 threads (8 waves). Tile 64(o) x 128(l), K-step 32. Wave w owns l-strip
// w and 4 o-subtiles. X tiles arrive via double-buffered TDM (f32 or f16 as
// stored; f16 conversion happens in the B-operand gather). W tiles staged by
// threads (f32 -> f16).
// ---------------------------------------------------------------------------
template <bool IN_F16, bool OUT_F16>
__global__ __launch_bounds__(256) void gemm_bias_kernel(
    const float* __restrict__ W, const void* __restrict__ Xv,
    const float* __restrict__ bias, void* __restrict__ Yv,
    int M, int K, int N) {
  constexpr int XS = 132;                       // padded LDS row stride (elems)
  constexpr int ES = IN_F16 ? 2 : 4;            // element size in bytes
  __shared__ __align__(16) char Xbuf[2][32 * XS * ES];  // TDM destination
  __shared__ _Float16 Wl[2][64 * 32];

  const int tid = threadIdx.x, lane = tid & 31, wave = tid >> 5;
  const int l0 = blockIdx.x * 128, o0 = blockIdx.y * 64, bb = blockIdx.z;
  const char* Xg = (const char*)Xv + (size_t)bb * K * N * ES;

  // TDM pad codes: f16 row = 256B = 64 DW -> interval code 5, pad 2 DW (code 1)
  //                f32 row = 512B = 128 DW -> interval code 6, pad 4 DW (code 3)
  constexpr unsigned ESC = IN_F16 ? 1u : 2u;
  constexpr unsigned PIC = IN_F16 ? 5u : 6u;
  constexpr unsigned PAC = IN_F16 ? 1u : 3u;

  auto issue_x = [&](int b2, int kt) {
    tdm_load_2d(lds_off_u32(&Xbuf[b2][0]),
                Xg + ((size_t)kt * N + l0) * ES, 128u, 32u,
                (unsigned long long)N, ESC, PIC, PAC);
  };
  auto stage_w = [&](int b2, int kt) {
    int o = tid >> 2, kc = (tid & 3) * 8;
    const float* ws = W + (size_t)(o0 + o) * K + kt + kc;
#pragma unroll
    for (int j = 0; j < 8; ++j) Wl[b2][o * 32 + kc + j] = (_Float16)ws[j];
  };

  if (wave == 0) issue_x(0, 0);
  stage_w(0, 0);

  v8f acc[4] = {v8f{}, v8f{}, v8f{}, v8f{}};
  int buf = 0;
  for (int kt = 0; kt < K; kt += 32) {
    if (wave == 0) __builtin_amdgcn_s_wait_tensorcnt(0);
    __syncthreads();  // Xbuf[buf] (TDM) and Wl[buf] ready for everyone

    if (kt + 32 < K) {  // start next tile while this one computes
      if (wave == 0) issue_x(buf ^ 1, kt + 32);
      stage_w(buf ^ 1, kt + 32);
    }

    // B operand: K=0..15 lanes 0-15 / K=16..31 lanes 16-31, N = lane%16
    v16h bv;
    const int khiB = (lane < 16) ? 0 : 16;
#pragma unroll
    for (int e = 0; e < 16; ++e) {
      int idx = (khiB + e) * XS + wave * 16 + (lane & 15);
      if constexpr (IN_F16)
        bv[e] = ((const _Float16*)&Xbuf[buf][0])[idx];
      else
        bv[e] = (_Float16)((const float*)&Xbuf[buf][0])[idx];
    }
    // A operand: lanes 0-15 K=0..15, lanes 16-31 K=16..31
    const int khiA = (lane < 16) ? 0 : 16;
#pragma unroll
    for (int ot = 0; ot < 4; ++ot) {
      v16h av;
#pragma unroll
      for (int e = 0; e < 16; ++e) {
        int kk = khiA + e;
        av[e] = Wl[buf][(ot * 16 + (lane & 15)) * 32 + kk];
      }
      acc[ot] = wmma16(av, bv, acc[ot]);
    }
    __syncthreads();  // everyone done with buf before it is refilled
    buf ^= 1;
  }

  // epilogue: C/D layout -> VGPR r = row (M=r, +8 for lanes 16-31), N=lane%16
  const int rofs = (lane < 16) ? 0 : 8;
#pragma unroll
  for (int ot = 0; ot < 4; ++ot) {
#pragma unroll
    for (int r = 0; r < 8; ++r) {
      int o = o0 + ot * 16 + r + rofs;
      int l = l0 + wave * 16 + (lane & 15);
      float val = acc[ot][r] + bias[o];
      if constexpr (OUT_F16)
        ((_Float16*)Yv)[((size_t)bb * M + o) * N + l] = (_Float16)val;
      else
        ((float*)Yv)[((size_t)bb * M + o) * N + l] = val;
    }
  }
}

// ---------------------------------------------------------------------------
// Fused attention per (b, head, 32-query tile).
//   Phase 1: S = scale * Q^T K (WMMA, key_dim 16 zero-padded to K=32) -> LDS
//   Phase 2: softmax rows in LDS (exp kept, 1/sum kept aside)
//   Phase 3: O^T = P @ V^T (WMMA, contraction 1024 split across wave pairs)
//   Epilogue: *1/rowsum + depthwise conv1d(k=3) PE + b_pe -> f16 y
// qkv layout: [b][head*64 + r][l] halves; r: 0-15 q, 16-31 k, 32-63 v.
// ---------------------------------------------------------------------------
#define SSTR 1028  // padded f32 stride (breaks 8-row bank aliasing)

__global__ __launch_bounds__(256) void attn_kernel(
    const _Float16* __restrict__ qkv, const float* __restrict__ w_pe,
    const float* __restrict__ b_pe, _Float16* __restrict__ y) {
  __shared__ float S[32 * SSTR];
  __shared__ float rowred[32 * 8];
  __shared__ float rowinv[32];
  __shared__ float Cpart[4 * 256];

  const int tid = threadIdx.x, lane = tid & 31, wave = tid >> 5;
  const int q0 = blockIdx.x * 32, h = blockIdx.y, bb = blockIdx.z;
  const _Float16* qh = qkv + ((size_t)(bb * NH_ + h) * 64) * L_;

  // ---- Phase 1: S tiles. Each wave: 2 i-tiles x 8 j-tiles.
  const int chi = (lane < 16) ? 0 : 8;
  v16h aq[2];
#pragma unroll
  for (int it = 0; it < 2; ++it) {
    int i = q0 + it * 16 + (lane & 15);
#pragma unroll
    for (int e = 0; e < 16; ++e)  // K>=16 pad (lanes 16-31)
      aq[it][e] = (lane < 16) ? qh[(size_t)e * L_ + i] : (_Float16)0.f;
  }
  for (int jt = wave; jt < 64; jt += 8) {
    const int j0 = jt * 16;
    v16h bk;
    if (lane < 16) {  // B K=c rows 0..15; lanes 16-31 cover K=16..31 -> 0 pad
#pragma unroll
      for (int e = 0; e < 16; ++e)
        bk[e] = qh[(size_t)(16 + e) * L_ + j0 + lane];
    } else {
#pragma unroll
      for (int e = 0; e < 16; ++e) bk[e] = (_Float16)0.f;
    }
#pragma unroll
    for (int it = 0; it < 2; ++it) {
      v8f s8 = {};
      s8 = wmma16(aq[it], bk, s8);
      const int rbase = it * 16 + ((lane < 16) ? 0 : 8);
#pragma unroll
      for (int r = 0; r < 8; ++r)
        S[(size_t)(rbase + r) * SSTR + j0 + (lane & 15)] = s8[r] * SCALE_;
    }
  }

  // Prefetch the V panel (64 KB) for phase 3 while softmax runs.
  {
    int row = tid >> 3, seg = (tid & 7) * 128;
    __builtin_prefetch(qh + (size_t)(32 + row) * L_ + seg, 0, 3);
  }
  __syncthreads();

  // ---- Phase 2: softmax. 8 threads per row, 128 keys each.
  {
    const int row = tid >> 3, sub = tid & 7;
    float* srow = &S[(size_t)row * SSTR + sub * 128];
    float m = -3.0e38f;
    for (int j = 0; j < 128; ++j) m = fmaxf(m, srow[j]);
    rowred[row * 8 + sub] = m;
    __syncthreads();
    float mall = rowred[row * 8];
    for (int t = 1; t < 8; ++t) mall = fmaxf(mall, rowred[row * 8 + t]);
    __syncthreads();
    float s = 0.f;
    for (int j = 0; j < 128; ++j) {
      float e = __expf(srow[j] - mall);
      srow[j] = e;
      s += e;
    }
    rowred[row * 8 + sub] = s;
    __syncthreads();
    if (sub == 0) {
      float tot = 0.f;
      for (int t = 0; t < 8; ++t) tot += rowred[row * 8 + t];
      rowinv[row] = 1.0f / tot;
    }
    __syncthreads();
  }

  // ---- Phase 3: O^T(32x32) = P(32x1024) @ V^T(1024x32).
  const int tw = wave & 3, khalf = wave >> 2;
  const int it = tw >> 1, ct = tw & 1;
  const int jlo = (lane < 16) ? 0 : 16;
  const _Float16* vrow0 = qh + (size_t)(32 + ct * 16 + (lane & 15)) * L_;
  v8f acc = {};
  for (int kt = khalf * 16; kt < khalf * 16 + 16; ++kt) {
    const int j0 = kt * 32;
    const float* sr = &S[(size_t)(it * 16 + (lane & 15)) * SSTR + j0];
    v16h a, bv;
#pragma unroll
    for (int e = 0; e < 16; ++e) {
      int jj = e + jlo;  // A K layout, f32 -> f16
      a[e] = (_Float16)sr[jj];
      bv[e] = vrow0[j0 + jlo + e];
    }
    acc = wmma16(a, bv, acc);
  }
  if (khalf == 1) {
#pragma unroll
    for (int r = 0; r < 8; ++r) Cpart[tw * 256 + r * 32 + lane] = acc[r];
  }
  __syncthreads();

  // ---- Epilogue on waves 0-3.
  if (khalf == 0) {
    const int c = ct * 16 + (lane & 15);
    const int d = h * HD_ + c;
    const float w0 = w_pe[d * 3 + 0], w1 = w_pe[d * 3 + 1],
                w2 = w_pe[d * 3 + 2], bp = b_pe[d];
    const _Float16* vr = qh + (size_t)(32 + c) * L_;
#pragma unroll
    for (int r = 0; r < 8; ++r) {
      const int il = it * 16 + r + ((lane < 16) ? 0 : 8);
      const int i = q0 + il;
      float o = (acc[r] + Cpart[tw * 256 + r * 32 + lane]) * rowinv[il];
      float vm = (i > 0) ? (float)vr[i - 1] : 0.f;
      float vc = (float)vr[i];
      float vp = (i < L_ - 1) ? (float)vr[i + 1] : 0.f;
      y[((size_t)(bb * DIM_ + d)) * L_ + i] =
          (_Float16)(o + w0 * vm + w1 * vc + w2 * vp + bp);
    }
  }
}

// ---------------------------------------------------------------------------
extern "C" void kernel_launch(void* const* d_in, const int* in_sizes, int n_in,
                              void* d_out, int out_size, void* d_ws,
                              size_t ws_size, hipStream_t stream) {
  const float* x      = (const float*)d_in[0];  // [B,256,1024]
  const float* w_qkv  = (const float*)d_in[1];  // [512,256]
  const float* b_qkv  = (const float*)d_in[2];  // [512]
  const float* w_pe   = (const float*)d_in[3];  // [256,1,3]
  const float* b_pe   = (const float*)d_in[4];  // [256]
  const float* w_proj = (const float*)d_in[5];  // [256,256]
  const float* b_proj = (const float*)d_in[6];  // [256]

  // workspace: qkv f16 (8 MB) | y f16 (4 MB)
  _Float16* qkv_ws = (_Float16*)d_ws;
  _Float16* y_ws   = qkv_ws + (size_t)B_ * H_ * L_;

  // 1) qkv = w_qkv @ x + b_qkv  (f32 in via TDM, f16 out)
  gemm_bias_kernel<false, true><<<dim3(L_ / 128, H_ / 64, B_), 256, 0, stream>>>(
      w_qkv, (const void*)x, b_qkv, (void*)qkv_ws, H_, DIM_, L_);

  // 2) fused attention + softmax + PE conv  (f16 -> f16)
  attn_kernel<<<dim3(L_ / 32, NH_, B_), 256, 0, stream>>>(qkv_ws, w_pe, b_pe,
                                                          y_ws);

  // 3) out = w_proj @ y + b_proj  (f16 in via TDM, f32 out)
  gemm_bias_kernel<true, false><<<dim3(L_ / 128, DIM_ / 64, B_), 256, 0,
                                  stream>>>(w_proj, (const void*)y_ws, b_proj,
                                            d_out, DIM_, DIM_, L_);
  (void)in_sizes; (void)n_in; (void)out_size; (void)ws_size;
}
